// PointNetSetAbstraction_30391188587132
// MI455X (gfx1250) — matrix operand order repaired and checked by reading, not compiled, not run on hardware
//
#include <hip/hip_runtime.h>
#include <hip/hip_bf16.h>
#include <stdint.h>

// ---------------------------------------------------------------------------
// PointNet++ Set Abstraction for MI455X (gfx1250, wave32, WMMA)
//   xyz:    [B=8, 3, N=16384] f32
//   points: [B, 3, N] f32
//   out:    concat( new_xyz [B,3,S=2048], new_points [B,128,S] ) f32
// Pipeline: FPS -> kNN(top-32) -> group/center -> (GEMM+BN+ReLU)x3 -> max_K
// GEMMs: v_wmma_f32_16x16x32_f16, N-blocked (one wave = 16 rows x all Cout),
// so each A fragment load feeds NT WMMAs against register/L0-resident B frags.
// ---------------------------------------------------------------------------

typedef __attribute__((ext_vector_type(16))) _Float16 v16h;
typedef __attribute__((ext_vector_type(8)))  _Float16 v8h;
typedef __attribute__((ext_vector_type(4)))  _Float16 v4h;
typedef __attribute__((ext_vector_type(8)))  float    v8f;

#define B_   8
#define N_   16384
#define S_   2048
#define K_   32
#define M_   (B_ * S_ * K_)      // 524288 rows for all GEMMs
#define FPS_BLOCK 1024
#define PPT (N_ / FPS_BLOCK)     // 16 points per thread in FPS

// -------------------------- small utility kernels --------------------------

__global__ void zero_f32(float* p, int n) {
    int i = blockIdx.x * blockDim.x + threadIdx.x;
    if (i < n) p[i] = 0.0f;
}

// ------------------------------- FPS ---------------------------------------
// One workgroup per batch. xyz (3*16384 f32 = 192KB) lives in LDS (<=320KB/WGP).
// distance[] stays in registers (16/thread). Argmax = wave shfl reduce + LDS.
__global__ __launch_bounds__(FPS_BLOCK)
void fps_kernel(const float* __restrict__ xyz,
                float* __restrict__ nx_ws,     // [B,S,3] for later stages
                float* __restrict__ out_xyz)   // d_out[0:B*3*S] transposed
{
    extern __shared__ float sm[];
    float* sx = sm;
    float* sy = sm + N_;
    float* sz = sm + 2 * N_;
    float* rv = sm + 3 * N_;          // 32 wave maxima
    int*   ri = (int*)(rv + 32);      // 32 wave argmax indices
    int*   scur = ri + 32;            // shared current centroid index

    const int b = blockIdx.x;
    const int tid = threadIdx.x;
    const int lane = tid & 31;
    const int wave = tid >> 5;

    const float* gx = xyz + (size_t)b * 3 * N_;
    const float* gy = gx + N_;
    const float* gz = gy + N_;
    for (int i = tid; i < N_; i += FPS_BLOCK) {
        sx[i] = gx[i]; sy[i] = gy[i]; sz[i] = gz[i];
    }
    if (tid == 0) *scur = 0;
    __syncthreads();

    float dist[PPT];
#pragma unroll
    for (int j = 0; j < PPT; ++j) dist[j] = 1e10f;

    for (int it = 0; it < S_; ++it) {
        const int cur = *scur;                       // uniform LDS broadcast
        const float cx = sx[cur], cy = sy[cur], cz = sz[cur];
        if (tid == 0) {
            float* nw = nx_ws + ((size_t)b * S_ + it) * 3;
            nw[0] = cx; nw[1] = cy; nw[2] = cz;
            float* ox = out_xyz + (size_t)b * 3 * S_;
            ox[0 * S_ + it] = cx; ox[1 * S_ + it] = cy; ox[2 * S_ + it] = cz;
        }
        float bv = -1.0f; int bi = 0x7fffffff;
#pragma unroll
        for (int j = 0; j < PPT; ++j) {
            const int n = tid + j * FPS_BLOCK;
            const float dx = sx[n] - cx, dy = sy[n] - cy, dz = sz[n] - cz;
            const float d = dx * dx + dy * dy + dz * dz;
            const float nd = fminf(dist[j], d);
            dist[j] = nd;
            if (nd > bv || (nd == bv && n < bi)) { bv = nd; bi = n; }
        }
#pragma unroll
        for (int off = 16; off > 0; off >>= 1) {
            const float ov = __shfl_xor(bv, off);
            const int   oi = __shfl_xor(bi, off);
            if (ov > bv || (ov == bv && oi < bi)) { bv = ov; bi = oi; }
        }
        if (lane == 0) { rv[wave] = bv; ri[wave] = bi; }
        __syncthreads();
        if (wave == 0) {
            float v2 = rv[lane]; int i2 = ri[lane];
#pragma unroll
            for (int off = 16; off > 0; off >>= 1) {
                const float ov = __shfl_xor(v2, off);
                const int   oi = __shfl_xor(i2, off);
                if (ov > v2 || (ov == v2 && oi < i2)) { v2 = ov; i2 = oi; }
            }
            if (lane == 0) *scur = i2;
        }
        __syncthreads();
    }
}

// ------------------------------- kNN ---------------------------------------
// One wave per centroid; K=32 candidates live one-per-lane. Replace-the-max
// insertion driven by ballot; wave argmax via butterfly shuffles.
__device__ __forceinline__ void wave_argmax(float& v, int& l) {
#pragma unroll
    for (int off = 16; off > 0; off >>= 1) {
        const float ov = __shfl_xor(v, off);
        const int   ol = __shfl_xor(l, off);
        if (ov > v || (ov == v && ol < l)) { v = ov; l = ol; }
    }
}

__global__ __launch_bounds__(256)
void knn_kernel(const float* __restrict__ xyz,
                const float* __restrict__ nx_ws,   // [B,S,3]
                int* __restrict__ gidx)            // [B,S,32]
{
    const int w = (blockIdx.x * blockDim.x + threadIdx.x) >> 5;
    const int lane = threadIdx.x & 31;
    if (w >= B_ * S_) return;
    const int b = w / S_;

    const float* px = xyz + (size_t)b * 3 * N_;
    const float* py = px + N_;
    const float* pz = py + N_;
    const float* c = nx_ws + (size_t)w * 3;
    const float cx = c[0], cy = c[1], cz = c[2];

    float cd; int ci;
    {
        const float dx = px[lane] - cx, dy = py[lane] - cy, dz = pz[lane] - cz;
        cd = dx * dx + dy * dy + dz * dz; ci = lane;
    }
    float thr = cd; int ml = lane;
    wave_argmax(thr, ml);

    for (int j = 32; j < N_; j += 32) {
        const int n = j + lane;
        const float dx = px[n] - cx, dy = py[n] - cy, dz = pz[n] - cz;
        const float d = dx * dx + dy * dy + dz * dz;
        unsigned mask = (unsigned)__ballot(d < thr);
        while (mask) {
            const int src = __ffs((int)mask) - 1;
            mask &= mask - 1;
            const float dj = __shfl(d, src);
            if (dj < thr) {                        // uniform branch
                if (lane == ml) { cd = dj; ci = j + src; }
                thr = cd; ml = lane;
                wave_argmax(thr, ml);
            }
        }
    }
    gidx[(size_t)w * 32 + lane] = ci;
}

// --------------------------- feature grouping -------------------------------
// feat[m, 0:6] = [xyz(idx)-centroid, points(idx)], zero-padded to Kpad=32, f16.
__global__ __launch_bounds__(256)
void build_feat(const float* __restrict__ xyz, const float* __restrict__ pts,
                const float* __restrict__ nx_ws, const int* __restrict__ gidx,
                _Float16* __restrict__ A)
{
    const size_t m = (size_t)blockIdx.x * blockDim.x + threadIdx.x;
    if (m >= (size_t)M_) return;
    const int s_global = (int)(m >> 5);            // b*S + s
    const int b = s_global / S_;
    const int idx = gidx[m];

    const float* bx = xyz + (size_t)b * 3 * N_;
    const float* bp = pts + (size_t)b * 3 * N_;
    const float* nw = nx_ws + (size_t)s_global * 3;

    const float fx = bx[idx]          - nw[0];
    const float fy = bx[N_ + idx]     - nw[1];
    const float fz = bx[2 * N_ + idx] - nw[2];
    const float p0 = bp[idx], p1 = bp[N_ + idx], p2 = bp[2 * N_ + idx];

    _Float16* out = A + m * 32;
    v8h c0 = { (_Float16)fx, (_Float16)fy, (_Float16)fz,
               (_Float16)p0, (_Float16)p1, (_Float16)p2,
               (_Float16)0.f, (_Float16)0.f };
    v8h z = {};
    *(v8h*)(out)      = c0;
    *(v8h*)(out + 8)  = z;
    *(v8h*)(out + 16) = z;
    *(v8h*)(out + 24) = z;
}

// --------------------------- weight pre-pack --------------------------------
// Pack W^T (B-matrix, K x N tiles) into per-lane WMMA fragment order so the
// GEMM does one contiguous 32B load per lane per (k-step, n-tile).
// B layout (16-bit, 32x16): lanes 0-15 hold K=0..7 and K=16..23,
// lanes 16-31 hold K=8..15 and K=24..31.  => element e of v16h at lane l: k = e + (e>=8)*8 + (l>=16)*8
__global__ __launch_bounds__(32)
void pack_weights(const float* __restrict__ w0, const float* __restrict__ w1,
                  const float* __restrict__ w2, _Float16* __restrict__ Bf)
{
    const int f = blockIdx.x;            // 0..27 global fragment id
    const int lane = threadIdx.x & 31;
    const float* W; int Cin, ntile, ks;
    if (f < 4)        { W = w0; Cin = 6;  ks = 0;          ntile = f; }
    else if (f < 12)  { W = w1; Cin = 64; ks = (f - 4) / 4; ntile = (f - 4) % 4; }
    else              { W = w2; Cin = 64; ks = (f - 12) / 8; ntile = (f - 12) % 8; }

    const int n = ntile * 16 + (lane & 15);
    const int koff = (lane >= 16) ? 8 : 0;
    _Float16* out = Bf + ((size_t)f * 32 + lane) * 16;
#pragma unroll
    for (int e = 0; e < 16; ++e) {
        const int k = ks * 32 + koff + e + ((e >= 8) ? 8 : 0);
        out[e] = (_Float16)((k < Cin) ? W[(size_t)n * Cin + k] : 0.0f);
    }
}

// ------------------------------ WMMA GEMM -----------------------------------
// C[M,Cout] = A[M,Kpad](f16) * B(packed frags, f16), f32 accumulate.
// N-blocked: one wave owns a 16-row stripe and ALL Cout columns (NT tiles),
// so each A fragment is loaded once per k-step and feeds NT WMMAs.
// Also accumulates per-channel sum/sumsq for training-mode BatchNorm.
// A layout (16-bit, 16x32): lane l holds row l%16; element e:
//   e<8 : k = e + (l>=16)*8 ; e>=8 : k = e + 8 + (l>=16)*8
// => two contiguous 8-half chunks at k0 = (l>=16)*8 and k0+16.
template <int KS, int NT>
__global__ __launch_bounds__(256)
void gemm_f16_wmma(const _Float16* __restrict__ A, const _Float16* __restrict__ Bf,
                   float* __restrict__ C, float* __restrict__ sum,
                   float* __restrict__ sumsq, int Mtiles)
{
    const int mtile = (blockIdx.x * blockDim.x + threadIdx.x) >> 5;
    const int lane = threadIdx.x & 31;
    if (mtile >= Mtiles) return;

    constexpr int Kpad = KS * 32;
    constexpr int Cout = NT * 16;

    const int row = mtile * 16 + (lane & 15);
    const int aoff = (lane >> 4) << 3;  // 0 or 8
    const _Float16* abase = A + (size_t)row * Kpad + aoff;

    v8f acc[NT];
#pragma unroll
    for (int nt = 0; nt < NT; ++nt) acc[nt] = (v8f){};

#pragma unroll
    for (int ks = 0; ks < KS; ++ks) {
        const v8h alo = *(const v8h*)(abase + ks * 32);
        const v8h ahi = *(const v8h*)(abase + ks * 32 + 16);
        v16h a;
#pragma unroll
        for (int i = 0; i < 8; ++i) { a[i] = alo[i]; a[i + 8] = ahi[i]; }
#pragma unroll
        for (int nt = 0; nt < NT; ++nt) {
            const v16h b =
                *(const v16h*)(Bf + ((size_t)(ks * NT + nt) * 32 + lane) * 16);
            acc[nt] = __builtin_amdgcn_wmma_f32_16x16x32_f16(
                false, a, false, b, (short)0, acc[nt], false, false);
        }
    }

    const int r0 = mtile * 16 + ((lane >> 4) << 3);
#pragma unroll
    for (int nt = 0; nt < NT; ++nt) {
        const int col = nt * 16 + (lane & 15);
        float s = 0.f, q = 0.f;
#pragma unroll
        for (int v = 0; v < 8; ++v) {
            const float x = acc[nt][v];
            C[(size_t)(r0 + v) * Cout + col] = x;
            s += x; q += x * x;
        }
        s += __shfl_xor(s, 16);   // lanes l and l^16 share column `col`
        q += __shfl_xor(q, 16);
        if (lane < 16) {
            atomicAdd(&sum[col], s);
            atomicAdd(&sumsq[col], q);
        }
    }
}

// --------------------------- BN fold / apply --------------------------------
__global__ void bn_finalize(const float* __restrict__ sum, const float* __restrict__ sumsq,
                            const float* __restrict__ g, const float* __restrict__ beta,
                            float* __restrict__ scale, float* __restrict__ shift,
                            int C, float inv_cnt)
{
    const int c = threadIdx.x;
    if (c < C) {
        const float mu = sum[c] * inv_cnt;
        const float var = sumsq[c] * inv_cnt - mu * mu;
        const float rs = rsqrtf(var + 1e-5f);
        const float sc = g[c] * rs;
        scale[c] = sc;
        shift[c] = beta[c] - mu * sc;
    }
}

// 4-wide BN+ReLU+f32->f16 (C divisible by 4)
__global__ __launch_bounds__(256)
void bn_relu_f16_v4(const float4* __restrict__ x, const float* __restrict__ scale,
                    const float* __restrict__ shift, _Float16* __restrict__ a,
                    size_t total4, int C4)
{
    const size_t i = (size_t)blockIdx.x * blockDim.x + threadIdx.x;
    if (i >= total4) return;
    const int c0 = (int)(i % C4) * 4;
    const float4 v = x[i];
    v4h o;
    o[0] = (_Float16)fmaxf(v.x * scale[c0 + 0] + shift[c0 + 0], 0.0f);
    o[1] = (_Float16)fmaxf(v.y * scale[c0 + 1] + shift[c0 + 1], 0.0f);
    o[2] = (_Float16)fmaxf(v.z * scale[c0 + 2] + shift[c0 + 2], 0.0f);
    o[3] = (_Float16)fmaxf(v.w * scale[c0 + 3] + shift[c0 + 3], 0.0f);
    *(v4h*)(a + i * 4) = o;
}

// ------------------- final BN + ReLU + max over K + transpose ---------------
__global__ __launch_bounds__(256)
void bn_relu_max_out(const float* __restrict__ x3, const float* __restrict__ scale,
                     const float* __restrict__ shift, float* __restrict__ outp)
{
    const size_t i = (size_t)blockIdx.x * blockDim.x + threadIdx.x;
    if (i >= (size_t)B_ * S_ * 128) return;
    const int c = (int)(i & 127);
    const int sg = (int)(i >> 7);           // b*S + s
    const int b = sg / S_;
    const int s = sg % S_;
    const float sc = scale[c], sh = shift[c];
    const float* p = x3 + (size_t)sg * K_ * 128 + c;
    float m = 0.0f;                          // relu floor
#pragma unroll
    for (int k = 0; k < K_; ++k)
        m = fmaxf(m, p[(size_t)k * 128] * sc + sh);
    outp[(size_t)b * 128 * S_ + (size_t)c * S_ + s] = m;
}

// ------------------------------- launcher -----------------------------------

extern "C" void kernel_launch(void* const* d_in, const int* in_sizes, int n_in,
                              void* d_out, int out_size, void* d_ws, size_t ws_size,
                              hipStream_t stream) {
    const float* xyz   = (const float*)d_in[0];
    const float* pts   = (const float*)d_in[1];
    const float* w0    = (const float*)d_in[2];
    const float* g0    = (const float*)d_in[4];
    const float* beta0 = (const float*)d_in[5];
    const float* w1    = (const float*)d_in[6];
    const float* g1    = (const float*)d_in[8];
    const float* beta1 = (const float*)d_in[9];
    const float* w2    = (const float*)d_in[10];
    const float* g2    = (const float*)d_in[12];
    const float* beta2 = (const float*)d_in[13];
    // Conv biases b0/b1/b2 cancel exactly inside training-mode BatchNorm
    // (constant per-channel shift removed by the mean) -> skipped.

    float* out_xyz = (float*)d_out;                       // [B,3,S]
    float* out_pts = (float*)d_out + (size_t)B_ * 3 * S_; // [B,128,S]

    // ---- workspace carve-up (~372 MB total) ----
    char* ws = (char*)d_ws;
    size_t off = 0;
    auto carve = [&](size_t bytes) -> char* {
        char* p = ws + off;
        off = (off + bytes + 255) & ~(size_t)255;
        return p;
    };
    float*    nx_ws  = (float*)   carve((size_t)B_ * S_ * 3 * sizeof(float));   // 0.19 MB
    int*      gidx   = (int*)     carve((size_t)B_ * S_ * 32 * sizeof(int));    // 2 MB
    _Float16* Bfrags = (_Float16*)carve((size_t)28 * 512 * sizeof(_Float16));   // 28 KB
    float*    stats  = (float*)   carve((size_t)3 * 512 * sizeof(float));       // 6 KB
    _Float16* featA  = (_Float16*)carve((size_t)M_ * 32 * sizeof(_Float16));    // 33.5 MB
    _Float16* act    = (_Float16*)carve((size_t)M_ * 64 * sizeof(_Float16));    // 67 MB
    float*    xbuf   = (float*)   carve((size_t)M_ * 128 * sizeof(float));      // 268 MB
    (void)ws_size; (void)out_size; (void)n_in; (void)in_sizes;

    // per-layer stats slices: [sum(128), sumsq(128), scale(128), shift(128)]
    float* sum0 = stats;        float* sq0 = stats + 128;
    float* sc0  = stats + 256;  float* sh0 = stats + 384;
    float* sum1 = stats + 512;  float* sq1 = stats + 640;
    float* sc1  = stats + 768;  float* sh1 = stats + 896;
    float* sum2 = stats + 1024; float* sq2 = stats + 1152;
    float* sc2  = stats + 1280; float* sh2 = stats + 1408;

    const float inv_cnt = 1.0f / (float)M_;
    const int Mtiles = M_ / 16;                 // 32768 (one wave each)
    const int gemm_blocks = Mtiles / 8;         // 8 waves / 256-thread block

    // 0) zero BN accumulators (deterministic each call)
    zero_f32<<<6, 256, 0, stream>>>(stats, 1536);

    // 1) pack weight fragments for WMMA B-operand
    pack_weights<<<28, 32, 0, stream>>>(w0, w1, w2, Bfrags);

    // 2) farthest point sampling (xyz resident in LDS)
    const size_t fps_lds = (size_t)(3 * N_ + 32) * sizeof(float) + 32 * sizeof(int) + 16;
    fps_kernel<<<B_, FPS_BLOCK, fps_lds, stream>>>(xyz, nx_ws, out_xyz);

    // 3) top-32 nearest neighbors (one wave per centroid)
    knn_kernel<<<(B_ * S_ * 32) / 256, 256, 0, stream>>>(xyz, nx_ws, gidx);

    // 4) gather + center + concat -> f16 features, K padded 6->32
    build_feat<<<M_ / 256, 256, 0, stream>>>(xyz, pts, nx_ws, gidx, featA);

    // 5) layer 1: [M,32pad] x [32,64] -> x1 (+stats)
    gemm_f16_wmma<1, 4><<<gemm_blocks, 256, 0, stream>>>(
        featA, Bfrags, xbuf, sum0, sq0, Mtiles);
    bn_finalize<<<1, 64, 0, stream>>>(sum0, sq0, g0, beta0, sc0, sh0, 64, inv_cnt);
    bn_relu_f16_v4<<<(unsigned)(((size_t)M_ * 64 / 4) / 256), 256, 0, stream>>>(
        (const float4*)xbuf, sc0, sh0, act, (size_t)M_ * 64 / 4, 16);

    // 6) layer 2: [M,64] x [64,64] -> x2 (+stats)
    gemm_f16_wmma<2, 4><<<gemm_blocks, 256, 0, stream>>>(
        act, Bfrags + (size_t)4 * 512, xbuf, sum1, sq1, Mtiles);
    bn_finalize<<<1, 64, 0, stream>>>(sum1, sq1, g1, beta1, sc1, sh1, 64, inv_cnt);
    bn_relu_f16_v4<<<(unsigned)(((size_t)M_ * 64 / 4) / 256), 256, 0, stream>>>(
        (const float4*)xbuf, sc1, sh1, act, (size_t)M_ * 64 / 4, 16);

    // 7) layer 3: [M,64] x [64,128] -> x3 (+stats)
    gemm_f16_wmma<2, 8><<<gemm_blocks, 256, 0, stream>>>(
        act, Bfrags + (size_t)12 * 512, xbuf, sum2, sq2, Mtiles);
    bn_finalize<<<1, 128, 0, stream>>>(sum2, sq2, g2, beta2, sc2, sh2, 128, inv_cnt);

    // 8) BN + ReLU + max over K=32 + transpose to [B,128,S]
    bn_relu_max_out<<<(unsigned)(((size_t)B_ * S_ * 128) / 256), 256, 0, stream>>>(
        xbuf, sc2, sh2, out_pts);
}
